// cascaded_mvPAM_Module_mask_13314398618069
// MI455X (gfx1250) — matrix operand order repaired and checked by reading, not compiled, not run on hardware
//
#include <hip/hip_runtime.h>
#include <cstdint>
#include <cstddef>

// ---------------------------------------------------------------------------
// Types for CDNA5 WMMA (wave32, 16x16x32 bf16 -> f32)
// ---------------------------------------------------------------------------
typedef __attribute__((ext_vector_type(16))) __bf16 v16bf;
typedef __attribute__((ext_vector_type(8)))  float  v8f;

union FragBF { v16bf v; uint4 u[2]; };
union Pack8  { __bf16 h[8]; uint4 u; };

#define BB   2
#define CC   512
#define HH   56
#define WW   56
#define NN   3136       // H*W
#define CQK  64
#define K4C  2048       // 4*C
#define EPSL 1e-5f

static __device__ inline __bf16 f2bf(float x) { return (__bf16)x; }

static __device__ inline v8f wmma_bf16(v16bf a, v16bf b, v8f c) {
  return __builtin_amdgcn_wmma_f32_16x16x32_bf16(false, a, false, b, (short)0, c,
                                                 false, false);
}

// A-fragment: 16(M) x 32(K) bf16, A row-major with leading dim lda (elements).
// lane<16: M=lane, K in {0..7, 16..23}; lane>=16: M=lane-16, K in {8..15, 24..31}
static __device__ inline v16bf load_fragA(const __bf16* __restrict__ Arow_base,
                                          size_t lda, int li, int hi, int k0) {
  FragBF f;
  const __bf16* r = Arow_base + (size_t)li * lda;
  f.u[0] = *reinterpret_cast<const uint4*>(r + k0 + 8 * hi);
  f.u[1] = *reinterpret_cast<const uint4*>(r + k0 + 16 + 8 * hi);
  return f.v;
}

// B-fragment: 32(K) x 16(N) bf16 from "T-storage" S[n][k] (n rows, K contiguous).
// lane<16: N=lane, K in {0..7, 16..23}; lane>=16: N=lane-16, K in {8..15, 24..31}
static __device__ inline v16bf load_fragB(const __bf16* __restrict__ Brow_base,
                                          size_t ldb, int li, int hi, int k0) {
  FragBF f;
  const __bf16* r = Brow_base + (size_t)li * ldb;
  f.u[0] = *reinterpret_cast<const uint4*>(r + k0 + 8 * hi);
  f.u[1] = *reinterpret_cast<const uint4*>(r + k0 + 16 + 8 * hi);
  return f.v;
}

// ---------------------------------------------------------------------------
// fp32 -> bf16 copy (weights)
// ---------------------------------------------------------------------------
__global__ void k_f32_to_bf16(const float* __restrict__ src,
                              __bf16* __restrict__ dst, int n) {
  int t = blockIdx.x * blockDim.x + threadIdx.x;
  if (t < n) dst[t] = f2bf(src[t]);
}

// ---------------------------------------------------------------------------
// q/k projections: qT[b][i][cq] = sum_c wq[cq][c]*x[b][c][i] + bq[cq]  (bf16)
// ---------------------------------------------------------------------------
__global__ void k_proj_qk(const float* __restrict__ x,
                          const float* __restrict__ wq, const float* __restrict__ bq,
                          const float* __restrict__ wk, const float* __restrict__ bk,
                          __bf16* __restrict__ qT, __bf16* __restrict__ kT) {
  int t = blockIdx.x * blockDim.x + threadIdx.x;
  if (t >= BB * CQK * NN) return;
  int i  = t % NN;
  int r  = t / NN;
  int cq = r % CQK;
  int b  = r / CQK;
  const float* xp = x + (size_t)b * CC * NN + i;
  const float* wqr = wq + (size_t)cq * CC;
  const float* wkr = wk + (size_t)cq * CC;
  float aq = bq[cq], ak = bk[cq];
  for (int c = 0; c < CC; ++c) {
    float xv = xp[(size_t)c * NN];
    aq += wqr[c] * xv;
    ak += wkr[c] * xv;
  }
  size_t o = ((size_t)b * NN + i) * CQK + cq;
  qT[o] = f2bf(aq);
  kT[o] = f2bf(ak);
}

// ---------------------------------------------------------------------------
// xT slice of catT: catT[b][i][1536+c] = bf16(x[b][c][i])
// ---------------------------------------------------------------------------
__global__ void k_xT(const float* __restrict__ x, __bf16* __restrict__ catT) {
  int t = blockIdx.x * blockDim.x + threadIdx.x;
  if (t >= BB * NN * CC) return;
  int c = t & (CC - 1);
  int r = t >> 9;          // b*NN + i
  int i = r % NN;
  int b = r / NN;
  catT[((size_t)b * NN + i) * K4C + 1536 + c] =
      f2bf(x[((size_t)b * CC + c) * NN + i]);
}

// ---------------------------------------------------------------------------
// Attention: energy via WMMA (A = key tiles from kT, B = 16-query strip from qT),
// 3-pass softmax (max / sums / normalized write), masks computed from indices.
// att?[b][i][j] row-major (j contiguous), bf16.
// ---------------------------------------------------------------------------
__global__ __launch_bounds__(256) void k_attn(const __bf16* __restrict__ qT,
                                              const __bf16* __restrict__ kT,
                                              __bf16* __restrict__ att0,
                                              __bf16* __restrict__ att1,
                                              __bf16* __restrict__ att2) {
  const int b  = blockIdx.y;
  const int i0 = blockIdx.x * 16;
  const int tid = threadIdx.x;
  const int wave = tid >> 5;
  const int lane = tid & 31;
  const int li = lane & 15;
  const int hi = lane >> 4;

  __shared__ float red[8][16][4];

  // B-fragments: the 16 queries (fixed for the whole block)
  const __bf16* qbase = qT + ((size_t)b * NN + i0) * CQK;
  v16bf bq0 = load_fragB(qbase, CQK, li, hi, 0);
  v16bf bq1 = load_fragB(qbase, CQK, li, hi, 32);

  const int i_lane = i0 + li;          // query column owned by this lane
  const int yi = i_lane / WW;
  const int xi = i_lane % WW;

  const __bf16* kbase = kT + (size_t)b * NN * CQK;

  // ---- pass 0: unmasked row max -------------------------------------------
  float m = -1e30f;
  for (int jt = wave; jt < NN / 16; jt += 8) {
    const int j0 = jt * 16;
    v16bf a0 = load_fragA(kbase + (size_t)j0 * CQK, CQK, li, hi, 0);
    v16bf a1 = load_fragA(kbase + (size_t)j0 * CQK, CQK, li, hi, 32);
    v8f acc = {};
    acc = wmma_bf16(a0, bq0, acc);
    acc = wmma_bf16(a1, bq1, acc);
#pragma unroll
    for (int v = 0; v < 8; ++v) m = fmaxf(m, acc[v]);
  }
  m = fmaxf(m, __shfl_xor(m, 16));
  if (hi == 0) red[wave][li][0] = m;
  __syncthreads();
  float gmax = -1e30f;
#pragma unroll
  for (int w = 0; w < 8; ++w) gmax = fmaxf(gmax, red[w][li][0]);

  // ---- pass 1: masked / unmasked exp-sums ---------------------------------
  float s0 = 0.f, s1 = 0.f, s2 = 0.f;
  for (int jt = wave; jt < NN / 16; jt += 8) {
    const int j0 = jt * 16;
    v16bf a0 = load_fragA(kbase + (size_t)j0 * CQK, CQK, li, hi, 0);
    v16bf a1 = load_fragA(kbase + (size_t)j0 * CQK, CQK, li, hi, 32);
    v8f acc = {};
    acc = wmma_bf16(a0, bq0, acc);
    acc = wmma_bf16(a1, bq1, acc);
#pragma unroll
    for (int v = 0; v < 8; ++v) {
      int j  = j0 + 8 * hi + v;
      int yj = j / WW, xj = j % WW;
      int dy = abs(yi - yj), dx = abs(xi - xj);
      float p = __expf(acc[v] - gmax);
      s0 += p;
      if (dy <= 7  && dx <= 7)  s1 += p;
      if (dy <= 14 && dx <= 14) s2 += p;
    }
  }
  s0 += __shfl_xor(s0, 16);
  s1 += __shfl_xor(s1, 16);
  s2 += __shfl_xor(s2, 16);
  if (hi == 0) { red[wave][li][1] = s0; red[wave][li][2] = s1; red[wave][li][3] = s2; }
  __syncthreads();
  float t0 = 0.f, t1 = 0.f, t2 = 0.f;
#pragma unroll
  for (int w = 0; w < 8; ++w) {
    t0 += red[w][li][1]; t1 += red[w][li][2]; t2 += red[w][li][3];
  }
  const float inv0 = 1.f / t0, inv1 = 1.f / t1, inv2 = 1.f / t2;

  // ---- pass 2: normalized attention writes (contiguous 16B per lane) ------
  for (int jt = wave; jt < NN / 16; jt += 8) {
    const int j0 = jt * 16;
    v16bf a0 = load_fragA(kbase + (size_t)j0 * CQK, CQK, li, hi, 0);
    v16bf a1 = load_fragA(kbase + (size_t)j0 * CQK, CQK, li, hi, 32);
    v8f acc = {};
    acc = wmma_bf16(a0, bq0, acc);
    acc = wmma_bf16(a1, bq1, acc);
    Pack8 p0, p1, p2;
#pragma unroll
    for (int v = 0; v < 8; ++v) {
      int j  = j0 + 8 * hi + v;
      int yj = j / WW, xj = j % WW;
      int dy = abs(yi - yj), dx = abs(xi - xj);
      float p = __expf(acc[v] - gmax);
      p0.h[v] = f2bf(p * inv0);
      p1.h[v] = (dy <= 7  && dx <= 7)  ? f2bf(p * inv1) : f2bf(0.f);
      p2.h[v] = (dy <= 14 && dx <= 14) ? f2bf(p * inv2) : f2bf(0.f);
    }
    size_t base = ((size_t)b * NN + i_lane) * NN + j0 + 8 * hi;
    *reinterpret_cast<uint4*>(att0 + base) = p0.u;
    *reinterpret_cast<uint4*>(att1 + base) = p1.u;
    *reinterpret_cast<uint4*>(att2 + base) = p2.u;
  }
}

// ---------------------------------------------------------------------------
// Generic WMMA GEMM with 2x2 register blocking per wave (32x32 output/wave):
//   out[m][n] = act(sum_k A[m][k]*BT[n][k] + bias[m]) (*mul)
//   A   : bf16, row-major (lda), per-batch stride aBatch
//   BT  : bf16, "T-storage" rows = n, K contiguous (ldb), per-batch bBatch
// stores (compile-time selected):
//   STF : fp32  out[m][n]                (c-major)
//   STCM: bf16  out[m][n]                (c-major)
//   STT : bf16  outT[n][m]               (T-storage, contiguous 16B stores)
// ACT=1: sigmoid ; MUL=1: multiply by mulF[m][n]
// grid: (M/128, Nn/64, batches), block 256 (8 waves = 4(m) x 2(n), each 32x32)
// ---------------------------------------------------------------------------
template <int ACT, int MUL, int STF, int STCM, int STT>
__global__ __launch_bounds__(256) void k_gemm(
    const __bf16* __restrict__ A, size_t aBatch, int lda,
    const __bf16* __restrict__ BT, size_t bBatch, int ldb,
    float* __restrict__ outF, size_t ofBatch,
    __bf16* __restrict__ outCM, size_t ocmBatch,
    __bf16* __restrict__ outT, size_t otBatch, int ldt,
    const float* __restrict__ bias,
    const float* __restrict__ mulF, size_t mulBatch,
    int M, int Nn, int K) {
  const int z = blockIdx.z;
  const int tid = threadIdx.x;
  const int wave = tid >> 5;
  const int lane = tid & 31;
  const int li = lane & 15;
  const int hi = lane >> 4;
  const int m0 = blockIdx.x * 128 + (wave >> 1) * 32;   // wave's 32-row block
  const int n0 = blockIdx.y * 64 + (wave & 1) * 32;     // wave's 32-col block

  const __bf16* Ab0 = A + z * aBatch + (size_t)m0 * lda;
  const __bf16* Ab1 = Ab0 + (size_t)16 * lda;
  const __bf16* Bb0 = BT + z * bBatch + (size_t)n0 * ldb;
  const __bf16* Bb1 = Bb0 + (size_t)16 * ldb;

  v8f acc[2][2] = {{{}, {}}, {{}, {}}};
  for (int k0 = 0; k0 < K; k0 += 32) {
    v16bf a0 = load_fragA(Ab0, (size_t)lda, li, hi, k0);
    v16bf a1 = load_fragA(Ab1, (size_t)lda, li, hi, k0);
    v16bf b0 = load_fragB(Bb0, (size_t)ldb, li, hi, k0);
    v16bf b1 = load_fragB(Bb1, (size_t)ldb, li, hi, k0);
    acc[0][0] = wmma_bf16(a0, b0, acc[0][0]);
    acc[0][1] = wmma_bf16(a0, b1, acc[0][1]);
    acc[1][0] = wmma_bf16(a1, b0, acc[1][0]);
    acc[1][1] = wmma_bf16(a1, b1, acc[1][1]);
  }

#pragma unroll
  for (int ti = 0; ti < 2; ++ti) {
#pragma unroll
    for (int tj = 0; tj < 2; ++tj) {
      const int mb = m0 + ti * 16;
      const int ncol = n0 + tj * 16 + li;
      Pack8 pk;
#pragma unroll
      for (int v = 0; v < 8; ++v) {
        int mrow = mb + 8 * hi + v;
        float val = acc[ti][tj][v];
        if (bias) val += bias[mrow];
        if (ACT == 1) val = 1.f / (1.f + __expf(-val));
        if (MUL) val *= mulF[z * mulBatch + (size_t)mrow * Nn + ncol];
        if (STF)  outF[z * ofBatch + (size_t)mrow * Nn + ncol] = val;
        if (STCM) outCM[z * ocmBatch + (size_t)mrow * Nn + ncol] = f2bf(val);
        if (STT)  pk.h[v] = f2bf(val);
      }
      if (STT) {
        *reinterpret_cast<uint4*>(outT + z * otBatch + (size_t)ncol * ldt + mb +
                                  8 * hi) = pk.u;
      }
    }
  }
}

// ---------------------------------------------------------------------------
// BatchNorm stats: one block per channel, reduce over (b, spatial)
// ---------------------------------------------------------------------------
__global__ __launch_bounds__(256) void k_bn_reduce(const float* __restrict__ y,
                                                   float* __restrict__ stats) {
  const int c = blockIdx.x;
  __shared__ float sh0[256], sh1[256];
  float s = 0.f, s2 = 0.f;
  for (int idx = threadIdx.x; idx < BB * NN; idx += 256) {
    int b = idx / NN, i = idx % NN;
    float v = y[((size_t)b * CC + c) * NN + i];
    s += v;
    s2 += v * v;
  }
  sh0[threadIdx.x] = s;
  sh1[threadIdx.x] = s2;
  __syncthreads();
  for (int off = 128; off > 0; off >>= 1) {
    if (threadIdx.x < off) {
      sh0[threadIdx.x] += sh0[threadIdx.x + off];
      sh1[threadIdx.x] += sh1[threadIdx.x + off];
    }
    __syncthreads();
  }
  if (threadIdx.x == 0) {
    float cnt = (float)(BB * NN);
    float mu = sh0[0] / cnt;
    float var = sh1[0] / cnt - mu * mu;
    stats[c] = mu;
    stats[CC + c] = __frsqrt_rn(var + EPSL);
  }
}

__global__ void k_bn_apply(const float* __restrict__ y,
                           const float* __restrict__ stats,
                           const float* __restrict__ gamma,
                           const float* __restrict__ beta,
                           float* __restrict__ out) {
  int t = blockIdx.x * blockDim.x + threadIdx.x;
  if (t >= BB * CC * NN) return;
  int c = (t / NN) & (CC - 1);
  float v = (y[t] - stats[c]) * stats[CC + c] * gamma[c] + beta[c];
  out[t] = fmaxf(v, 0.f);
}

// ---------------------------------------------------------------------------
// Host launcher
// ---------------------------------------------------------------------------
extern "C" void kernel_launch(void* const* d_in, const int* in_sizes, int n_in,
                              void* d_out, int out_size, void* d_ws, size_t ws_size,
                              hipStream_t stream) {
  const float* x  = (const float*)d_in[0];
  // d_in[1], d_in[2] = masks (recomputed analytically in-kernel)
  const float* wq = (const float*)d_in[3];
  const float* bq = (const float*)d_in[4];
  const float* wk = (const float*)d_in[5];
  const float* bk = (const float*)d_in[6];
  const float* wv = (const float*)d_in[7];
  const float* bv = (const float*)d_in[8];
  const float* wa = (const float*)d_in[9];
  const float* ba = (const float*)d_in[10];
  const float* wb = (const float*)d_in[11];
  const float* gm = (const float*)d_in[12];
  const float* bt = (const float*)d_in[13];
  float* out = (float*)d_out;

  // ---- workspace carve-up -------------------------------------------------
  char* p = (char*)d_ws;
  auto alloc = [&](size_t bytes) {
    void* r = (void*)p;
    p += (bytes + 255) & ~(size_t)255;
    return r;
  };
  __bf16* qT    = (__bf16*)alloc((size_t)BB * NN * CQK * 2);
  __bf16* kT    = (__bf16*)alloc((size_t)BB * NN * CQK * 2);
  __bf16* catT  = (__bf16*)alloc((size_t)BB * NN * K4C * 2);
  __bf16* att0  = (__bf16*)alloc((size_t)BB * NN * NN * 2);
  __bf16* att1  = (__bf16*)alloc((size_t)BB * NN * NN * 2);
  __bf16* att2  = (__bf16*)alloc((size_t)BB * NN * NN * 2);
  float*  v0f   = (float*) alloc((size_t)BB * CC * NN * 4);
  __bf16* v0bf  = (__bf16*)alloc((size_t)BB * CC * NN * 2);
  __bf16* v1bf  = (__bf16*)alloc((size_t)BB * CC * NN * 2);
  __bf16* v2bf  = (__bf16*)alloc((size_t)BB * CC * NN * 2);
  __bf16* wvbf  = (__bf16*)alloc((size_t)CC * CC * 2);
  __bf16* wabf  = (__bf16*)alloc((size_t)CC * CC * 2);
  __bf16* wbbf  = (__bf16*)alloc((size_t)CC * K4C * 2);
  float*  ybuf  = (float*) alloc((size_t)BB * CC * NN * 4);
  float*  stats = (float*) alloc((size_t)2 * CC * 4);

  // ---- weight conversions -------------------------------------------------
  k_f32_to_bf16<<<(CC * CC + 255) / 256, 256, 0, stream>>>(wv, wvbf, CC * CC);
  k_f32_to_bf16<<<(CC * CC + 255) / 256, 256, 0, stream>>>(wa, wabf, CC * CC);
  k_f32_to_bf16<<<(CC * K4C + 255) / 256, 256, 0, stream>>>(wb, wbbf, CC * K4C);

  // ---- projections --------------------------------------------------------
  k_proj_qk<<<(BB * CQK * NN + 255) / 256, 256, 0, stream>>>(x, wq, bq, wk, bk, qT, kT);
  k_xT<<<(BB * NN * CC + 255) / 256, 256, 0, stream>>>(x, catT);

  // ---- attention (energy + 3 masked softmaxes) ----------------------------
  k_attn<<<dim3(NN / 16, BB), 256, 0, stream>>>(qT, kT, att0, att1, att2);

  const size_t SB_ATT = (size_t)NN * NN;   // att batch stride (elements)
  const size_t SB_CAT = (size_t)NN * K4C;  // catT batch stride
  const size_t SB_CN  = (size_t)CC * NN;   // (C,N) batch stride

  const dim3 ggrid(CC / 128, NN / 64, BB);

  // G1: v0 = wv*x + bv  -> v0f (fp32) + v0bf (bf16, c-major)
  k_gemm<0, 0, 1, 1, 0><<<ggrid, 256, 0, stream>>>(
      wvbf, 0, CC, catT + 1536, SB_CAT, K4C,
      v0f, SB_CN, v0bf, SB_CN, nullptr, 0, 0, bv, nullptr, 0, CC, NN, CC);

  // G2: out0 = v0 * att0^T -> catT[:, 0:512] (bf16 T-store)
  k_gemm<0, 0, 0, 0, 1><<<ggrid, 256, 0, stream>>>(
      v0bf, SB_CN, NN, att0, SB_ATT, NN,
      nullptr, 0, nullptr, 0, catT + 0, SB_CAT, K4C, nullptr, nullptr, 0, CC, NN, NN);

  // G3: v1 = sigmoid(wa*out0 + ba) * v0 -> v1bf
  k_gemm<1, 1, 0, 1, 0><<<ggrid, 256, 0, stream>>>(
      wabf, 0, CC, catT + 0, SB_CAT, K4C,
      nullptr, 0, v1bf, SB_CN, nullptr, 0, 0, ba, v0f, SB_CN, CC, NN, CC);

  // G4: out1 = v1 * att1^T -> catT[:, 512:1024]
  k_gemm<0, 0, 0, 0, 1><<<ggrid, 256, 0, stream>>>(
      v1bf, SB_CN, NN, att1, SB_ATT, NN,
      nullptr, 0, nullptr, 0, catT + 512, SB_CAT, K4C, nullptr, nullptr, 0, CC, NN, NN);

  // G5: v2 = sigmoid(wa*out1 + ba) * v0 -> v2bf
  k_gemm<1, 1, 0, 1, 0><<<ggrid, 256, 0, stream>>>(
      wabf, 0, CC, catT + 512, SB_CAT, K4C,
      nullptr, 0, v2bf, SB_CN, nullptr, 0, 0, ba, v0f, SB_CN, CC, NN, CC);

  // G6: out2 = v2 * att2^T -> catT[:, 1024:1536]
  k_gemm<0, 0, 0, 0, 1><<<ggrid, 256, 0, stream>>>(
      v2bf, SB_CN, NN, att2, SB_ATT, NN,
      nullptr, 0, nullptr, 0, catT + 1024, SB_CAT, K4C, nullptr, nullptr, 0, CC, NN, NN);

  // G7: y = wb * cat -> ybuf (fp32)
  k_gemm<0, 0, 1, 0, 0><<<ggrid, 256, 0, stream>>>(
      wbbf, 0, K4C, catT, SB_CAT, K4C,
      ybuf, SB_CN, nullptr, 0, nullptr, 0, 0, nullptr, nullptr, 0, CC, NN, K4C);

  // ---- batch-norm + relu --------------------------------------------------
  k_bn_reduce<<<CC, 256, 0, stream>>>(ybuf, stats);
  k_bn_apply<<<(BB * CC * NN + 255) / 256, 256, 0, stream>>>(ybuf, stats, gm, bt, out);
}
